// GsLstm_41437844471984
// MI455X (gfx1250) — matrix operand order repaired and checked by reading, not compiled, not run on hardware
//
#include <hip/hip_runtime.h>
#include <hip/hip_bf16.h>

// Problem constants (reference setup_inputs)
#define Bc 8
#define Nc 2000
#define Hc 128
#define Kc 32
#define Mc (Bc * Nc)      // 16000 rows
#define KT 384            // combined K: [in_agg | out_agg | h]
#define DT 512            // 4 gates x 128
#define KSTEPS (KT / 32)  // 12
#define WSLICE (DT * 32)  // 16384 bf16 = 32 KB per K-step slice
#define LAYERS 2

#if __has_builtin(__builtin_amdgcn_global_load_async_to_lds_b128) && \
    __has_builtin(__builtin_amdgcn_s_wait_asynccnt)
#define ASYNC_LDS 1
#endif

typedef __attribute__((ext_vector_type(16))) __bf16 v16bf;
typedef __attribute__((ext_vector_type(8)))  float v8f;

// Pointer types for the async-to-LDS builtin (per hipcc diagnostic: the
// builtin takes pointers to 16-byte int vectors in AS1 / AS3).
typedef int v4i_ __attribute__((vector_size(4 * sizeof(int))));
typedef __attribute__((address_space(1))) v4i_ gv4;
typedef __attribute__((address_space(3))) v4i_ lv4;

union Frag16 { v16bf v; unsigned int u[8]; uint4 q[2]; };

__device__ __forceinline__ unsigned short f2bf(float x) {
    unsigned int u = __float_as_uint(x);
    unsigned int r = u + 0x7FFFu + ((u >> 16) & 1u);   // round-to-nearest-even
    return (unsigned short)(r >> 16);
}

// ---------------------------------------------------------------------------
// Build combined bf16 weight matrix, stored in per-K-step fragment layout:
//   WcT[step][col][klocal]  with k = step*32 + klocal, col = g*128 + d
//   k in [0,128):   w_in[g][k][d]
//   k in [128,256): w_out[g][k-128][d]
//   k in [256,384): u_in[g][k-256][d] + u_out[g][k-256][d]
// ---------------------------------------------------------------------------
__global__ void prep_weights(const float* __restrict__ w_in,
                             const float* __restrict__ w_out,
                             const float* __restrict__ u_in,
                             const float* __restrict__ u_out,
                             unsigned short* __restrict__ WcT) {
    int idx = blockIdx.x * blockDim.x + threadIdx.x;
    if (idx >= KT * DT) return;
    int k = idx >> 9;          // / 512
    int col = idx & (DT - 1);
    int g = col >> 7;
    int d = col & (Hc - 1);
    float v;
    if (k < Hc) {
        v = w_in[(g * Hc + k) * Hc + d];
    } else if (k < 2 * Hc) {
        v = w_out[(g * Hc + (k - Hc)) * Hc + d];
    } else {
        int kk = k - 2 * Hc;
        v = u_in[(g * Hc + kk) * Hc + d] + u_out[(g * Hc + kk) * Hc + d];
    }
    WcT[(size_t)(k >> 5) * WSLICE + col * 32 + (k & 31)] = f2bf(v);
}

// ---------------------------------------------------------------------------
// Neighbor aggregation: one block per node (row = b*N + n), 128 threads = H.
// Writes bf16 combined A row: [in_agg(128) | out_agg(128) | h(128)].
// ---------------------------------------------------------------------------
__global__ void agg_kernel(const float* __restrict__ h,
                           const int*   __restrict__ inIdx,
                           const float* __restrict__ inMsk,
                           const int*   __restrict__ outIdx,
                           const float* __restrict__ outMsk,
                           unsigned short* __restrict__ Abuf) {
    __shared__ int   sIn[Kc];
    __shared__ float sInM[Kc];
    __shared__ int   sOut[Kc];
    __shared__ float sOutM[Kc];

    const int row = blockIdx.x;          // b*N + n
    const int b   = row / Nc;
    const int tid = threadIdx.x;         // 0..127 (H index)

    if (tid < Kc) {
        int base = row * Kc + tid;
        sIn[tid]   = inIdx[base];
        sInM[tid]  = inMsk[base];
        sOut[tid]  = outIdx[base];
        sOutM[tid] = outMsk[base];
    }
    __syncthreads();

    const float* hb = h + (size_t)b * Nc * Hc;
    float accIn = 0.f, accOut = 0.f;
#pragma unroll 4
    for (int k = 0; k < Kc; ++k) {
        accIn  += sInM[k]  * hb[(size_t)sIn[k]  * Hc + tid];
        accOut += sOutM[k] * hb[(size_t)sOut[k] * Hc + tid];
    }
    unsigned short* arow = Abuf + (size_t)row * KT;
    arow[tid]            = f2bf(accIn);
    arow[Hc + tid]       = f2bf(accOut);
    arow[2 * Hc + tid]   = f2bf(h[(size_t)row * Hc + tid]);
}

// ---------------------------------------------------------------------------
// Stage one 32KB W slice into LDS. Async DMA path when available (overlaps
// with WMMA of the current step); synchronous uint4 copy otherwise.
// ---------------------------------------------------------------------------
__device__ __forceinline__ void stage_wslice(const unsigned short* __restrict__ WcT,
                                             unsigned short* dst, int step, int tid) {
#ifdef ASYNC_LDS
    gv4* gsrc = (gv4*)(WcT + (size_t)step * WSLICE + tid * 8);  // 16B per lane
    lv4* ldst = (lv4*)(dst + tid * 8);
#pragma unroll
    for (int i = 0; i < 8; ++i) {
        // 256 lanes x 16B per pass = 4096B; 8 passes = 32KB. Offset folded
        // into the pointers (builtin offset must be a literal constant).
        __builtin_amdgcn_global_load_async_to_lds_b128(gsrc + i * 512,
                                                       ldst + i * 512, 0, 0);
    }
#else
    for (int j = tid; j < WSLICE / 8; j += 256)
        *(uint4*)&dst[j * 8] = *(const uint4*)&WcT[(size_t)step * WSLICE + j * 8];
#endif
}

__device__ __forceinline__ void wait_async() {
#ifdef ASYNC_LDS
    __builtin_amdgcn_s_wait_asynccnt(0);
#endif
}

// ---------------------------------------------------------------------------
// GEMM + fused LSTM gates.
// Block: 256 threads (8 wave32 waves), tile = 32 rows x 512 cols.
// Wave w: rows (w&1)*16..+15, gate g = w>>1 (128 cols) = 8 WMMA 16x16 tiles.
// K = 384 swept in 12 steps of 32 with v_wmma_f32_16x16x32_bf16; next W slice
// is DMA'd into the alternate LDS buffer while the current step computes.
// All 8 B fragments are loaded first (two ds_load_b128 each), then the 8
// WMMAs issue back-to-back on disjoint accumulators (no D->A/B hazards).
// ---------------------------------------------------------------------------
__global__ void __launch_bounds__(256)
gemm_gates(const unsigned short* __restrict__ Abuf,
           const unsigned short* __restrict__ WcT,
           const float* __restrict__ bias,
           float* __restrict__ cbuf,
           float* __restrict__ hbuf,
           float* __restrict__ hout) {
    __shared__ union {
        struct {
            unsigned short As[32 * KT];      // 24 KB: A tile (32 rows x 384)
            unsigned short Wst[2][WSLICE];   // 2 x 32 KB: W slice double buffer
        } s;
        float pre[32 * DT];                  // 64 KB: gate pre-acts (overlay)
    } sm;

    const int tid     = threadIdx.x;
    const int lane    = tid & 31;
    const int wave    = tid >> 5;
    const int half    = lane >> 4;         // 0: lanes 0-15, 1: lanes 16-31
    const int nlane   = lane & 15;
    const int rBlock  = wave & 1;          // row strip within 32-row tile
    const int g       = wave >> 1;         // gate / 128-col strip
    const int colBase = g * Hc;
    const int rowBase = blockIdx.x * 32;
    const int r16     = rBlock * 16;

    // Kick off first W slice DMA; stage A tile (32 x 384 bf16) meanwhile.
    stage_wslice(WcT, sm.s.Wst[0], 0, tid);
    for (int j = tid; j < (32 * KT) / 8; j += 256) {   // 1536 uint4 chunks
        int row = j / 48;
        int off = (j % 48) * 8;
        *(uint4*)&sm.s.As[row * KT + off] =
            *(const uint4*)&Abuf[(size_t)(rowBase + row) * KT + off];
    }
    wait_async();
    __syncthreads();

    v8f acc[8];
#pragma unroll
    for (int t = 0; t < 8; ++t) acc[t] = v8f{0.f,0.f,0.f,0.f,0.f,0.f,0.f,0.f};

    for (int ks = 0; ks < KSTEPS; ++ks) {
        const unsigned short* wbuf = sm.s.Wst[ks & 1];
        if (ks + 1 < KSTEPS)               // prefetch next slice during compute
            stage_wslice(WcT, sm.s.Wst[(ks + 1) & 1], ks + 1, tid);

        const int k0 = ks * 32;
        // A fragment: per-lane data is two contiguous 16B runs
        // (ushorts [k0+half*8, +8) and [k0+16+half*8, +8)) -> 2x b128 loads.
        Frag16 a;
        {
            const unsigned short* ap = &sm.s.As[(r16 + nlane) * KT + k0 + half * 8];
            a.q[0] = *(const uint4*)(ap);
            a.q[1] = *(const uint4*)(ap + 16);
        }
        // B fragments: per-lane data is two 16B runs at
        // wbuf[col*32 + half*8] and +16 -> 2x b128 loads each. Load all 8 first.
        Frag16 bf[8];
#pragma unroll
        for (int t = 0; t < 8; ++t) {
            const unsigned short* bp =
                &wbuf[(colBase + t * 16 + nlane) * 32 + half * 8];
            bf[t].q[0] = *(const uint4*)(bp);
            bf[t].q[1] = *(const uint4*)(bp + 16);
        }
        // 8 WMMAs back-to-back on disjoint accumulators.
#pragma unroll
        for (int t = 0; t < 8; ++t) {
            acc[t] = __builtin_amdgcn_wmma_f32_16x16x32_bf16(
                false, a.v, false, bf[t].v, (short)0, acc[t], false, false);
        }
        wait_async();
        __syncthreads();
    }

    // Spill accumulators into the overlaid fp32 pre buffer (D layout: M = r + 8*half).
#pragma unroll
    for (int t = 0; t < 8; ++t) {
#pragma unroll
        for (int r = 0; r < 8; ++r) {
            int rowL = r16 + r + 8 * half;
            sm.pre[rowL * DT + colBase + t * 16 + nlane] = acc[t][r];
        }
    }
    __syncthreads();

    // Fused gates: i,o,f,g -> cell/hidden update.
    for (int e = tid; e < 32 * Hc; e += 256) {
        int row = e >> 7;
        int d   = e & (Hc - 1);
        float iv = sm.pre[row * DT + d]            + bias[d];
        float ov = sm.pre[row * DT + Hc + d]       + bias[Hc + d];
        float fv = sm.pre[row * DT + 2 * Hc + d]   + bias[2 * Hc + d];
        float gv = sm.pre[row * DT + 3 * Hc + d]   + bias[3 * Hc + d];
        float si = 1.f / (1.f + __expf(-iv));
        float so = 1.f / (1.f + __expf(-ov));
        float sf = 1.f / (1.f + __expf(-fv));
        float gg = tanhf(gv);
        size_t gr = (size_t)(rowBase + row) * Hc + d;
        float cn = sf * cbuf[gr] + si * gg;
        float hn = so * tanhf(cn);
        cbuf[gr] = cn;
        hbuf[gr] = hn;
        if (hout) hout[gr] = hn;
    }
}

// ---------------------------------------------------------------------------
extern "C" void kernel_launch(void* const* d_in, const int* in_sizes, int n_in,
                              void* d_out, int out_size, void* d_ws, size_t ws_size,
                              hipStream_t stream) {
    const float* node_hidden = (const float*)d_in[0];
    const float* cell        = (const float*)d_in[1];
    const float* w_in        = (const float*)d_in[2];
    const float* w_out       = (const float*)d_in[3];
    const float* u_in        = (const float*)d_in[4];
    const float* u_out       = (const float*)d_in[5];
    const float* bias        = (const float*)d_in[6];
    const float* in_msk      = (const float*)d_in[7];
    const float* out_msk     = (const float*)d_in[8];
    const int*   in_idx      = (const int*)d_in[9];
    const int*   out_idx     = (const int*)d_in[10];

    const size_t stateBytes = (size_t)Mc * Hc * sizeof(float);          // 8,192,000
    const size_t abufBytes  = (size_t)Mc * KT * sizeof(unsigned short); // 12,288,000

    char* ws = (char*)d_ws;
    float*          hbuf = (float*)ws;
    float*          cbuf = (float*)(ws + stateBytes);
    unsigned short* Abuf = (unsigned short*)(ws + 2 * stateBytes);
    unsigned short* WcT  = (unsigned short*)(ws + 2 * stateBytes + abufBytes);

    (void)hipMemcpyAsync(hbuf, node_hidden, stateBytes, hipMemcpyDeviceToDevice, stream);
    (void)hipMemcpyAsync(cbuf, cell,        stateBytes, hipMemcpyDeviceToDevice, stream);

    prep_weights<<<(KT * DT + 255) / 256, 256, 0, stream>>>(w_in, w_out, u_in, u_out, WcT);

    for (int layer = 0; layer < LAYERS; ++layer) {
        agg_kernel<<<Mc, Hc, 0, stream>>>(hbuf, in_idx, in_msk, out_idx, out_msk, Abuf);
        gemm_gates<<<Mc / 32, 256, 0, stream>>>(
            Abuf, WcT, bias, cbuf, hbuf,
            (layer == LAYERS - 1) ? (float*)d_out : nullptr);
    }
}
